// SACB_57543971832453
// MI455X (gfx1250) — matrix operand order repaired and checked by reading, not compiled, not run on hardware
//
#include <hip/hip_runtime.h>
#include <math.h>

typedef __attribute__((ext_vector_type(16))) _Float16 v16h;
typedef __attribute__((ext_vector_type(8)))  _Float16 v8h;
typedef __attribute__((ext_vector_type(8)))  float    v8f;

#define DIMV 64
#define CH   16
#define NVOX (DIMV*DIMV*DIMV)   // 262144
#define K27  27
#define KTOT (CH*K27)           // 432
#define NTAP 28                 // 27 real taps + 1 zero pad tap
#define KPAD (NTAP*CH)          // 448, K order: kk = tap*16 + channel
#define NKSTEP (KPAD/32)        // 14
#define NUMK 4
#define MH1 128
#define MH2 64
#define NROWS (DIMV*DIMV)       // 4096 (z,y) rows of 64 voxels
#define XS   20                 // padded channel stride in input tile (16B aligned, bank-friendly)

// ---------------------------------------------------------------------------
// WMMA GEMM core used by both conv kernels.
// Input tile ytf: [(dz*3+dy)*66 + xx][c] with channel stride XS (f32).
// Weight panel At: [o][tap*16 + c] (f16, tap 27 zeroed).
// Each wave: 16 voxels along x; lane fragment uses contiguous k = 16*hi + s for A and B.
// B element build: lane tap = 2*ks + hi; 16 contiguous f32 -> 4x ds_load_b128,
// scaled by one scalar factor (modulation or 1.0; 0.0 for the pad tap).
// ---------------------------------------------------------------------------

// Kernel 1: proj conv 3x3x3 (16->16) + bias, per-row InstanceNorm partials
__global__ __launch_bounds__(128)
void k_conv_proj(const float* __restrict__ x, const float* __restrict__ pw,
                 const float* __restrict__ pb, float* __restrict__ y,
                 float* __restrict__ psum, float* __restrict__ psq)
{
    __shared__ __align__(16) float    ytf[3*3*66*XS];
    __shared__ __align__(32) _Float16 At[CH][KPAD];
    __shared__ int   tapb[NTAP];
    __shared__ float wsum[4][16];
    __shared__ float wsq [4][16];

    const int tid = threadIdx.x;
    const int row = blockIdx.x;
    const int zz  = row >> 6;
    const int yy  = row & 63;
    const int n0  = row << 6;

    // stage zero-padded input tile, channel-innermost
    for (int idx = tid; idx < CH*3*3*66; idx += 128) {
        int c  = idx / (3*3*66);
        int r  = idx % (3*3*66);
        int dz = r / (3*66);
        int r2 = r % (3*66);
        int dy = r2 / 66;
        int xx = r2 % 66;
        int gz = zz + dz - 1, gy = yy + dy - 1, gx = xx - 1;
        float v = 0.f;
        if ((unsigned)gz < DIMV && (unsigned)gy < DIMV && (unsigned)gx < DIMV)
            v = x[((c*DIMV + gz)*DIMV + gy)*DIMV + gx];
        ytf[((dz*3 + dy)*66 + xx)*XS + c] = v;
    }
    // stage weights permuted to tap-major K order, f16, pad tap zeroed
    for (int idx = tid; idx < CH*KPAD; idx += 128) {
        int o = idx / KPAD, kk = idx % KPAD;
        int k = kk >> 4, c = kk & 15;
        At[o][kk] = (k < K27) ? (_Float16)pw[o*KTOT + c*K27 + k] : (_Float16)0.f;
    }
    // tap -> LDS base offset table (pad tap aliases tap 0; zeroed via factor)
    if (tid < NTAP) {
        int k = (tid < K27) ? tid : 0;
        int dz = k/9, r = k%9, dy = r/3, dx = r%3;
        tapb[tid] = ((dz*3 + dy)*66 + dx)*XS;
    }
    __syncthreads();

    const int lane = tid & 31;
    const int wave = tid >> 5;
    const int nl   = lane & 15;
    const int hi   = lane >> 4;
    const int xv   = wave*16 + nl;

    v8f acc = {0.f,0.f,0.f,0.f,0.f,0.f,0.f,0.f};
    #pragma unroll
    for (int ks = 0; ks < NKSTEP; ++ks) {
        const int kb = ks*32;
        union { v16h v; v8h h[2]; } au;
        au.h[0] = *(const v8h*)&At[nl][kb + hi*16];
        au.h[1] = *(const v8h*)&At[nl][kb + hi*16 + 8];

        const int kl = 2*ks + hi;                       // this lane's tap
        const float mf = (kl < K27) ? 1.f : 0.f;        // zero the pad tap
        const float4* bp = (const float4*)&ytf[tapb[kl] + xv*XS];
        float4 b0 = bp[0], b1 = bp[1], b2 = bp[2], b3 = bp[3];
        union { v16h v; _Float16 e[16]; } bu;
        bu.e[0]=(_Float16)(b0.x*mf); bu.e[1]=(_Float16)(b0.y*mf);
        bu.e[2]=(_Float16)(b0.z*mf); bu.e[3]=(_Float16)(b0.w*mf);
        bu.e[4]=(_Float16)(b1.x*mf); bu.e[5]=(_Float16)(b1.y*mf);
        bu.e[6]=(_Float16)(b1.z*mf); bu.e[7]=(_Float16)(b1.w*mf);
        bu.e[8]=(_Float16)(b2.x*mf); bu.e[9]=(_Float16)(b2.y*mf);
        bu.e[10]=(_Float16)(b2.z*mf); bu.e[11]=(_Float16)(b2.w*mf);
        bu.e[12]=(_Float16)(b3.x*mf); bu.e[13]=(_Float16)(b3.y*mf);
        bu.e[14]=(_Float16)(b3.z*mf); bu.e[15]=(_Float16)(b3.w*mf);

        acc = __builtin_amdgcn_wmma_f32_16x16x32_f16(
                  false, au.v, false, bu.v, (short)0, acc, false, false);
    }

    // epilogue: bias, store y, deterministic per-row channel sum/sumsq
    const int n = n0 + xv;
    float s[8], q[8];
    #pragma unroll
    for (int i = 0; i < 8; ++i) {
        int o = hi*8 + i;
        float val = acc[i] + pb[o];
        y[o*NVOX + n] = val;
        s[i] = val; q[i] = val*val;
    }
    #pragma unroll
    for (int m = 8; m >= 1; m >>= 1) {
        #pragma unroll
        for (int i = 0; i < 8; ++i) {
            s[i] += __shfl_xor(s[i], m, 32);
            q[i] += __shfl_xor(q[i], m, 32);
        }
    }
    if (nl == 0) {
        #pragma unroll
        for (int i = 0; i < 8; ++i) {
            wsum[wave][hi*8+i] = s[i];
            wsq [wave][hi*8+i] = q[i];
        }
    }
    __syncthreads();
    if (tid < 16) {
        float ts = 0.f, tq = 0.f;
        for (int w = 0; w < 4; ++w) { ts += wsum[w][tid]; tq += wsq[w][tid]; }
        psum[row*16 + tid] = ts;
        psq [row*16 + tid] = tq;
    }
}

// final InstanceNorm stats (serial, deterministic)
__global__ void k_norm_stats(const float* __restrict__ psum,
                             const float* __restrict__ psq,
                             float* __restrict__ stats)
{
    int t = threadIdx.x;
    if (t < 16) {
        float s = 0.f, q = 0.f;
        for (int b = 0; b < NROWS; ++b) { s += psum[b*16+t]; q += psq[b*16+t]; }
        float mu  = s / (float)NVOX;
        float var = q / (float)NVOX - mu*mu;
        stats[t]      = mu;
        stats[16 + t] = rsqrtf(var + 1e-5f);
    }
}

// normalize + PReLU in place
__global__ __launch_bounds__(256)
void k_normalize(float* __restrict__ y, const float* __restrict__ stats,
                 const float* __restrict__ a_in)
{
    int i = blockIdx.x*256 + threadIdx.x;
    int c = i >> 18;
    float mu = stats[c], inv = stats[16+c], a = a_in[0];
    float v = (y[i] - mu) * inv;
    y[i] = (v >= 0.f) ? v : a*v;
}

// 27-tap zero-padded box mean -> x_mean in [c][n] (SoA) layout
__global__ __launch_bounds__(256)
void k_boxmean(const float* __restrict__ y, float* __restrict__ xmean)
{
    int n  = blockIdx.x*256 + threadIdx.x;
    int xn = n & 63, yn = (n >> 6) & 63, zn = n >> 12;
    for (int c = 0; c < CH; ++c) {
        float s = 0.f;
        for (int dz = -1; dz <= 1; ++dz)
        for (int dy = -1; dy <= 1; ++dy)
        for (int dx = -1; dx <= 1; ++dx) {
            int gz = zn+dz, gy = yn+dy, gx = xn+dx;
            if ((unsigned)gz < 64u && (unsigned)gy < 64u && (unsigned)gx < 64u)
                s += y[((c*64 + gz)*64 + gy)*64 + gx];
        }
        xmean[c*NVOX + n] = s * (1.f/27.f);
    }
}

// k-means deterministic init: pts at linspace(0,N-1,4) = {0,87381,174762,262143}
__global__ void k_km_init(const float* __restrict__ xmean, float* __restrict__ cent)
{
    int t = threadIdx.x;
    if (t < 64) {
        int q = t >> 4, c = t & 15;
        const int idxs[4] = {0, 87381, 174762, 262143};
        cent[q*16 + c] = xmean[c*NVOX + idxs[q]];
    }
}

// one Lloyd accumulate pass: 64 blocks x 256 thr, 16 pts/thread, no atomics
__global__ __launch_bounds__(256)
void k_km_acc(const float* __restrict__ xmean, const float* __restrict__ cent,
              float* __restrict__ part)
{
    __shared__ float cs[64];
    __shared__ float wbuf[8][68];
    int tid = threadIdx.x;
    if (tid < 64) cs[tid] = cent[tid];
    __syncthreads();

    float s[NUMK][16];
    float cnt[NUMK];
    #pragma unroll
    for (int q = 0; q < NUMK; ++q) {
        cnt[q] = 0.f;
        #pragma unroll
        for (int c = 0; c < 16; ++c) s[q][c] = 0.f;
    }
    const int base = blockIdx.x * 4096;
    for (int it = 0; it < 16; ++it) {
        int n = base + it*256 + tid;
        float p[16];
        #pragma unroll
        for (int c = 0; c < 16; ++c) p[c] = xmean[c*NVOX + n];
        float bd = 3.4e38f; int best = 0;
        #pragma unroll
        for (int q = 0; q < NUMK; ++q) {
            float d = 0.f;
            #pragma unroll
            for (int c = 0; c < 16; ++c) { float t2 = p[c]-cs[q*16+c]; d += t2*t2; }
            if (d < bd) { bd = d; best = q; }     // strict < => first-min wins
        }
        #pragma unroll
        for (int q = 0; q < NUMK; ++q) {
            float m = (q == best) ? 1.f : 0.f;
            cnt[q] += m;
            #pragma unroll
            for (int c = 0; c < 16; ++c) s[q][c] += m * p[c];
        }
    }
    #pragma unroll
    for (int msk = 16; msk >= 1; msk >>= 1) {
        #pragma unroll
        for (int q = 0; q < NUMK; ++q) {
            cnt[q] += __shfl_xor(cnt[q], msk, 32);
            #pragma unroll
            for (int c = 0; c < 16; ++c) s[q][c] += __shfl_xor(s[q][c], msk, 32);
        }
    }
    int lane = tid & 31, wave = tid >> 5;
    if (lane == 0) {
        for (int q = 0; q < NUMK; ++q) {
            for (int c = 0; c < 16; ++c) wbuf[wave][q*16+c] = s[q][c];
            wbuf[wave][64+q] = cnt[q];
        }
    }
    __syncthreads();
    if (tid < 68) {
        float t2 = 0.f;
        for (int w = 0; w < 8; ++w) t2 += wbuf[w][tid];
        part[blockIdx.x*68 + tid] = t2;
    }
}

__global__ void k_km_update(const float* __restrict__ part, float* __restrict__ cent)
{
    int t = threadIdx.x;
    if (t < 64) {
        int q = t >> 4;
        float sum = 0.f, cq = 0.f;
        for (int b = 0; b < 64; ++b) { sum += part[b*68 + t]; cq += part[b*68 + 64 + q]; }
        float old = cent[t];
        cent[t] = (cq > 0.f) ? sum / fmaxf(cq, 1.f) : old;
    }
}

__global__ __launch_bounds__(256)
void k_assign(const float* __restrict__ xmean, const float* __restrict__ cent,
              int* __restrict__ assign)
{
    __shared__ float cs[64];
    int tid = threadIdx.x;
    if (tid < 64) cs[tid] = cent[tid];
    __syncthreads();
    int n = blockIdx.x*256 + tid;
    float p[16];
    #pragma unroll
    for (int c = 0; c < 16; ++c) p[c] = xmean[c*NVOX + n];
    float bd = 3.4e38f; int best = 0;
    #pragma unroll
    for (int q = 0; q < NUMK; ++q) {
        float d = 0.f;
        #pragma unroll
        for (int c = 0; c < 16; ++c) { float t2 = p[c]-cs[q*16+c]; d += t2*t2; }
        if (d < bd) { bd = d; best = q; }
    }
    assign[n] = best;
}

// tiny centroid MLPs: mod = sigmoid(16->128->128->27), bvec = 16->64->64->16
__global__ __launch_bounds__(128)
void k_mlp(const float* __restrict__ cent,
           const float* kw1, const float* kb1, const float* kw2, const float* kb2,
           const float* kw3, const float* kb3,
           const float* bw1, const float* bb1, const float* bw2, const float* bb2,
           const float* bw3, const float* bb3,
           float* __restrict__ mod, float* __restrict__ bvec)
{
    __shared__ float cq[16];
    __shared__ float h1[MH1], h2[MH1], g1[MH2], g2[MH2];
    int t = threadIdx.x;
    for (int q = 0; q < NUMK; ++q) {
        if (t < 16) cq[t] = cent[q*16 + t];
        __syncthreads();
        { float z = kb1[t];
          for (int c = 0; c < 16; ++c) z += cq[c]*kw1[t*16 + c];
          h1[t] = fmaxf(z, 0.f); }
        if (t < MH2) {
            float z = bb1[t];
            for (int c = 0; c < 16; ++c) z += cq[c]*bw1[t*16 + c];
            g1[t] = fmaxf(z, 0.f);
        }
        __syncthreads();
        { float z = kb2[t];
          for (int j = 0; j < MH1; ++j) z += h1[j]*kw2[t*MH1 + j];
          h2[t] = fmaxf(z, 0.f); }
        if (t < MH2) {
            float z = bb2[t];
            for (int j = 0; j < MH2; ++j) z += g1[j]*bw2[t*MH2 + j];
            g2[t] = fmaxf(z, 0.f);
        }
        __syncthreads();
        if (t < K27) {
            float z = kb3[t];
            for (int j = 0; j < MH1; ++j) z += h2[j]*kw3[t*MH1 + j];
            mod[q*K27 + t] = 1.f / (1.f + expf(-z));
        }
        if (t < 16) {
            float z = bb3[t];
            for (int j = 0; j < MH2; ++j) z += g2[j]*bw3[t*MH2 + j];
            bvec[q*16 + t] = z;
        }
        __syncthreads();
    }
}

// ---------------------------------------------------------------------------
// Kernel 8: modulated conv WMMA GEMM + cluster bias + PReLU + residual
// ---------------------------------------------------------------------------
__global__ __launch_bounds__(128)
void k_main(const float* __restrict__ y, const float* __restrict__ w,
            const int* __restrict__ assign, const float* __restrict__ mod,
            const float* __restrict__ bvec, const float* __restrict__ a_out,
            const float* __restrict__ x, float* __restrict__ out)
{
    __shared__ __align__(16) float    ytf[3*3*66*XS];
    __shared__ __align__(32) _Float16 At[CH][KPAD];
    __shared__ int   tapb[NTAP];
    __shared__ float modx[NUMK][NTAP];   // per-cluster per-tap factor, pad tap = 0
    __shared__ float bvsh[NUMK][16];

    const int tid = threadIdx.x;
    const int row = blockIdx.x;
    const int zz  = row >> 6;
    const int yy  = row & 63;
    const int n0  = row << 6;

    for (int idx = tid; idx < CH*3*3*66; idx += 128) {
        int c  = idx / (3*3*66);
        int r  = idx % (3*3*66);
        int dz = r / (3*66);
        int r2 = r % (3*66);
        int dy = r2 / 66;
        int xx = r2 % 66;
        int gz = zz + dz - 1, gy = yy + dy - 1, gx = xx - 1;
        float v = 0.f;
        if ((unsigned)gz < DIMV && (unsigned)gy < DIMV && (unsigned)gx < DIMV)
            v = y[((c*DIMV + gz)*DIMV + gy)*DIMV + gx];
        ytf[((dz*3 + dy)*66 + xx)*XS + c] = v;
    }
    for (int idx = tid; idx < CH*KPAD; idx += 128) {
        int o = idx / KPAD, kk = idx % KPAD;
        int k = kk >> 4, c = kk & 15;
        At[o][kk] = (k < K27) ? (_Float16)w[o*KTOT + c*K27 + k] : (_Float16)0.f;
    }
    if (tid < NTAP) {
        int k = (tid < K27) ? tid : 0;
        int dz = k/9, r = k%9, dy = r/3, dx = r%3;
        tapb[tid] = ((dz*3 + dy)*66 + dx)*XS;
    }
    if (tid < NUMK*NTAP) {
        int q = tid / NTAP, k = tid % NTAP;
        modx[q][k] = (k < K27) ? mod[q*K27 + k] : 0.f;
    }
    if (tid < NUMK*16) bvsh[tid>>4][tid&15] = bvec[tid];
    __syncthreads();

    const int lane = tid & 31;
    const int wave = tid >> 5;
    const int nl   = lane & 15;
    const int hi   = lane >> 4;
    const int xv   = wave*16 + nl;
    const int n    = n0 + xv;
    const int av   = assign[n];

    v8f acc = {0.f,0.f,0.f,0.f,0.f,0.f,0.f,0.f};
    #pragma unroll
    for (int ks = 0; ks < NKSTEP; ++ks) {
        const int kb = ks*32;
        union { v16h v; v8h h[2]; } au;
        au.h[0] = *(const v8h*)&At[nl][kb + hi*16];
        au.h[1] = *(const v8h*)&At[nl][kb + hi*16 + 8];

        const int kl = 2*ks + hi;                  // this lane's tap
        const float mf = modx[av][kl];             // per-voxel-cluster tap scale
        const float4* bp = (const float4*)&ytf[tapb[kl] + xv*XS];
        float4 b0 = bp[0], b1 = bp[1], b2 = bp[2], b3 = bp[3];
        union { v16h v; _Float16 e[16]; } bu;
        bu.e[0]=(_Float16)(b0.x*mf); bu.e[1]=(_Float16)(b0.y*mf);
        bu.e[2]=(_Float16)(b0.z*mf); bu.e[3]=(_Float16)(b0.w*mf);
        bu.e[4]=(_Float16)(b1.x*mf); bu.e[5]=(_Float16)(b1.y*mf);
        bu.e[6]=(_Float16)(b1.z*mf); bu.e[7]=(_Float16)(b1.w*mf);
        bu.e[8]=(_Float16)(b2.x*mf); bu.e[9]=(_Float16)(b2.y*mf);
        bu.e[10]=(_Float16)(b2.z*mf); bu.e[11]=(_Float16)(b2.w*mf);
        bu.e[12]=(_Float16)(b3.x*mf); bu.e[13]=(_Float16)(b3.y*mf);
        bu.e[14]=(_Float16)(b3.z*mf); bu.e[15]=(_Float16)(b3.w*mf);

        acc = __builtin_amdgcn_wmma_f32_16x16x32_f16(
                  false, au.v, false, bu.v, (short)0, acc, false, false);
    }

    const float ao = a_out[0];
    #pragma unroll
    for (int i = 0; i < 8; ++i) {
        int o = hi*8 + i;
        float val = acc[i] + bvsh[av][o];
        val = (val >= 0.f) ? val : ao*val;          // PReLU
        val += x[o*NVOX + n];                       // residual
        out[o*NVOX + n] = val;
    }
}

// ---------------------------------------------------------------------------
extern "C" void kernel_launch(void* const* d_in, const int* in_sizes, int n_in,
                              void* d_out, int out_size, void* d_ws, size_t ws_size,
                              hipStream_t stream)
{
    const float* x      = (const float*)d_in[0];
    const float* proj_w = (const float*)d_in[1];
    const float* proj_b = (const float*)d_in[2];
    const float* a_in   = (const float*)d_in[3];
    const float* w      = (const float*)d_in[4];
    const float* kw1    = (const float*)d_in[5];
    const float* kb1    = (const float*)d_in[6];
    const float* kw2    = (const float*)d_in[7];
    const float* kb2    = (const float*)d_in[8];
    const float* kw3    = (const float*)d_in[9];
    const float* kb3    = (const float*)d_in[10];
    const float* bw1    = (const float*)d_in[11];
    const float* bb1    = (const float*)d_in[12];
    const float* bw2    = (const float*)d_in[13];
    const float* bb2    = (const float*)d_in[14];
    const float* bw3    = (const float*)d_in[15];
    const float* bb3    = (const float*)d_in[16];
    const float* a_out  = (const float*)d_in[17];
    float* out = (float*)d_out;

    char* ws = (char*)d_ws;
    float* y      = (float*)(ws);                                  // 16*N
    float* xmean  = (float*)(ws + (size_t)CH*NVOX*4);              // 16*N
    float* psum   = (float*)(ws + (size_t)CH*NVOX*8);              // 4096*16
    float* psq    = (float*)((char*)psum + (size_t)NROWS*16*4);
    float* stats  = (float*)((char*)psq  + (size_t)NROWS*16*4);    // 32
    float* cent   = (float*)((char*)stats + 32*4);                 // 64
    float* kmpart = (float*)((char*)cent  + 64*4);                 // 64*68
    int*   assign = (int*)  ((char*)kmpart + 64*68*4);             // N
    float* modb   = (float*)((char*)assign + (size_t)NVOX*4);      // 4*27
    float* bvecb  = (float*)((char*)modb   + 128*4);               // 4*16

    k_conv_proj<<<NROWS, 128, 0, stream>>>(x, proj_w, proj_b, y, psum, psq);
    k_norm_stats<<<1, 64, 0, stream>>>(psum, psq, stats);
    k_normalize<<<(CH*NVOX)/256, 256, 0, stream>>>(y, stats, a_in);
    k_boxmean<<<NVOX/256, 256, 0, stream>>>(y, xmean);
    k_km_init<<<1, 64, 0, stream>>>(xmean, cent);
    for (int it = 0; it < 15; ++it) {
        k_km_acc<<<64, 256, 0, stream>>>(xmean, cent, kmpart);
        k_km_update<<<1, 64, 0, stream>>>(kmpart, cent);
    }
    k_assign<<<NVOX/256, 256, 0, stream>>>(xmean, cent, assign);
    k_mlp<<<1, 128, 0, stream>>>(cent, kw1, kb1, kw2, kb2, kw3, kb3,
                                 bw1, bb1, bw2, bb2, bw3, bb3, modb, bvecb);
    k_main<<<NROWS, 128, 0, stream>>>(y, w, assign, modb, bvecb, a_out, x, out);
    (void)in_sizes; (void)n_in; (void)out_size; (void)ws_size;
}
